// MixtureOfExperts_66486093742329
// MI455X (gfx1250) — matrix operand order repaired and checked by reading, not compiled, not run on hardware
//
#include <hip/hip_runtime.h>
#include <cstdint>
#include <cstddef>

typedef _Float16 v16h __attribute__((ext_vector_type(16)));
typedef _Float16 v8h  __attribute__((ext_vector_type(8)));
typedef _Float16 v4h  __attribute__((ext_vector_type(4)));
typedef float    v8f  __attribute__((ext_vector_type(8)));
typedef float    v4f  __attribute__((ext_vector_type(4)));

#define B_BATCH 4
#define T_SEQ   1024
#define C_DIM   768
#define H_NUM   12
#define E_NUM   4
#define L_NUM   2
#define DH      64
#define BT      (B_BATCH * T_SEQ)   // 4096 tokens

// ---------------------------------------------------------------------------
// helpers
// ---------------------------------------------------------------------------
static __device__ __forceinline__ v16h cat8(v8h lo, v8h hi) {
  return __builtin_shufflevector(lo, hi, 0,1,2,3,4,5,6,7,8,9,10,11,12,13,14,15);
}

// tanh-GELU. gfx1250 has a native v_tanh_f32 transcendental; if the builtin
// is not exposed, use the exact branch-free identity 0.5x(1+tanh(u)) ==
// x * sigmoid(2u) == x / (1 + exp(-2u))  ->  v_exp_f32 + v_rcp_f32.
static __device__ __forceinline__ float gelu_tanh(float x) {
  float u = 0.7978845608028654f * (x + 0.044715f * x * x * x);
#if __has_builtin(__builtin_amdgcn_tanhf)
  return 0.5f * x * (1.0f + __builtin_amdgcn_tanhf(u));
#else
  return x / (1.0f + __expf(-2.0f * u));
#endif
}

// ---------------------------------------------------------------------------
// Gate: softmax(x @ gate_W + gate_b) over E=4 experts, one block per token
// ---------------------------------------------------------------------------
__global__ __launch_bounds__(256) void gate_kernel(
    const float* __restrict__ x, const float* __restrict__ gw,
    const float* __restrict__ gb, float* __restrict__ gate)
{
  __shared__ float red[E_NUM][256];
  const int row = blockIdx.x, tid = threadIdx.x;
  const float* xp = x + (size_t)row * C_DIM;
  float s[E_NUM] = {0.f, 0.f, 0.f, 0.f};
  for (int c = tid; c < C_DIM; c += 256) {
    float xv = xp[c];
    const v4f gwv = *reinterpret_cast<const v4f*>(&gw[(size_t)c * E_NUM]); // E=4, 16B rows
    #pragma unroll
    for (int e = 0; e < E_NUM; ++e) s[e] += xv * gwv[e];
  }
  #pragma unroll
  for (int e = 0; e < E_NUM; ++e) red[e][tid] = s[e];
  __syncthreads();
  for (int st = 128; st > 0; st >>= 1) {
    if (tid < st) {
      #pragma unroll
      for (int e = 0; e < E_NUM; ++e) red[e][tid] += red[e][tid + st];
    }
    __syncthreads();
  }
  if (tid == 0) {
    float lg[E_NUM], mx = -1e30f;
    #pragma unroll
    for (int e = 0; e < E_NUM; ++e) { lg[e] = red[e][0] + gb[e]; mx = fmaxf(mx, lg[e]); }
    float den = 0.f;
    #pragma unroll
    for (int e = 0; e < E_NUM; ++e) { lg[e] = __expf(lg[e] - mx); den += lg[e]; }
    float rd = 1.f / den;
    #pragma unroll
    for (int e = 0; e < E_NUM; ++e) gate[(size_t)row * E_NUM + e] = lg[e] * rd;
  }
}

// ---------------------------------------------------------------------------
// Generic WMMA GEMM:  D[M,N] = act(A[M,K] @ W[K,N] + bias[N])
// fp32 in/out, f16 WMMA with f32 accumulation.
// Block: 256 threads = 8 waves, tile 128x128, K-step 32.
// Waves: 4 along M x 2 along N; each wave: 32x64 = 2x4 WMMA accumulators.
// Software-pipelined: next K-tile global loads are issued right after the
// current tile's LDS stores and consumed one iteration later, so global
// latency hides behind the 8 WMMAs + fragment ds_loads of each step.
// ---------------------------------------------------------------------------
#define BM 128
#define BN 128
#define BK 32
#define LPAD 40   // 32 + 8 halves padding (80B rows, 16B aligned)

__global__ __launch_bounds__(256) void gemm_bias_act_kernel(
    const float* __restrict__ A, const float* __restrict__ W,
    const float* __restrict__ bias, float* __restrict__ D,
    int M, int N, int K, int act)
{
  __shared__ __attribute__((aligned(16))) _Float16 As[BM][LPAD];
  __shared__ __attribute__((aligned(16))) _Float16 Bs[BN][LPAD];  // transposed: [n][k]

  const int tid  = threadIdx.x;
  const int lane = tid & 31;
  const int wv   = tid >> 5;
  const int hi   = lane >> 4;      // half-wave select
  const int l16  = lane & 15;
  const int wm   = wv & 3;         // wave row (4 along M)
  const int wn   = wv >> 2;        // wave col (2 along N)
  const int bm   = blockIdx.x * BM;
  const int bn   = blockIdx.y * BN;

  // per-thread staging coordinates (float4 granularity), loop-invariant
  const int aR = tid >> 3;               // A tile row for it-chunk base (+32 per it)
  const int aC = (tid & 7) * 4;          // A tile col
  const int wK = tid >> 5;               // W tile k-row base (+8 per it)
  const int wN = (tid & 31) * 4;         // W tile col

  const v8f vzero = {0.f,0.f,0.f,0.f,0.f,0.f,0.f,0.f};
  v8f acc[2][4];
  #pragma unroll
  for (int i = 0; i < 2; ++i)
    #pragma unroll
    for (int j = 0; j < 4; ++j) acc[i][j] = vzero;

  // pipeline registers + prologue loads (k0 = 0)
  v4f aReg[4], wReg[4];
  #pragma unroll
  for (int it = 0; it < 4; ++it)
    aReg[it] = *reinterpret_cast<const v4f*>(&A[(size_t)(bm + aR + it * 32) * K + aC]);
  #pragma unroll
  for (int it = 0; it < 4; ++it)
    wReg[it] = *reinterpret_cast<const v4f*>(&W[(size_t)(wK + it * 8) * N + (bn + wN)]);

  for (int k0 = 0; k0 < K; k0 += BK) {
    __syncthreads();   // all waves done reading LDS from the previous step
    // commit staged tile to LDS (f32 -> f16)
    #pragma unroll
    for (int it = 0; it < 4; ++it) {
      v4h hv = { (_Float16)aReg[it][0], (_Float16)aReg[it][1],
                 (_Float16)aReg[it][2], (_Float16)aReg[it][3] };
      *reinterpret_cast<v4h*>(&As[aR + it * 32][aC]) = hv;
    }
    #pragma unroll
    for (int it = 0; it < 4; ++it) {
      Bs[wN + 0][wK + it * 8] = (_Float16)wReg[it][0];
      Bs[wN + 1][wK + it * 8] = (_Float16)wReg[it][1];
      Bs[wN + 2][wK + it * 8] = (_Float16)wReg[it][2];
      Bs[wN + 3][wK + it * 8] = (_Float16)wReg[it][3];
    }
    // issue next tile's global loads; waited on at next iteration's stores
    if (k0 + BK < K) {
      const int kn = k0 + BK;
      #pragma unroll
      for (int it = 0; it < 4; ++it)
        aReg[it] = *reinterpret_cast<const v4f*>(&A[(size_t)(bm + aR + it * 32) * K + (kn + aC)]);
      #pragma unroll
      for (int it = 0; it < 4; ++it)
        wReg[it] = *reinterpret_cast<const v4f*>(&W[(size_t)(kn + wK + it * 8) * N + (bn + wN)]);
    }
    __syncthreads();   // LDS tile visible to all waves

    v16h af[2], bf[4];
    #pragma unroll
    for (int mf = 0; mf < 2; ++mf) {
      int m = wm * 32 + mf * 16 + l16;
      const v8h* p = (const v8h*)&As[m][hi * 8];
      const v8h* q = (const v8h*)&As[m][16 + hi * 8];
      af[mf] = cat8(*p, *q);
    }
    #pragma unroll
    for (int nf = 0; nf < 4; ++nf) {
      int n = wn * 64 + nf * 16 + l16;
      const v8h* p = (const v8h*)&Bs[n][hi * 8];
      const v8h* q = (const v8h*)&Bs[n][16 + hi * 8];
      bf[nf] = cat8(*p, *q);
    }
    #pragma unroll
    for (int mf = 0; mf < 2; ++mf)
      #pragma unroll
      for (int nf = 0; nf < 4; ++nf)
        acc[mf][nf] = __builtin_amdgcn_wmma_f32_16x16x32_f16(
            false, af[mf], false, bf[nf], (short)0, acc[mf][nf], false, false);
  }

  // epilogue: bias (+ GELU) and store
  #pragma unroll
  for (int mf = 0; mf < 2; ++mf)
    #pragma unroll
    for (int nf = 0; nf < 4; ++nf)
      #pragma unroll
      for (int r = 0; r < 8; ++r) {
        int m = bm + wm * 32 + mf * 16 + r + 8 * hi;
        int n = bn + wn * 64 + nf * 16 + l16;
        float xv = acc[mf][nf][r] + bias[n];
        if (act) xv = gelu_tanh(xv);
        D[(size_t)m * N + n] = xv;
      }
}

// ---------------------------------------------------------------------------
// Flash attention (exact, unscaled, unmasked softmax(Q K^T) V per head).
// q,k,v,o: [B*T, C] fp32, head h occupies columns [h*64, h*64+64).
// Block: 128 threads = 4 waves; each wave owns a 16-query block. All waves
// share each 32-key K/V block, so K (row-major) and V (transposed) are staged
// cooperatively in LDS once per block: 4x less global traffic and all WMMA
// fragments become pairs of 16B ds_loads.
// ---------------------------------------------------------------------------
__global__ __launch_bounds__(128) void flash_attn_kernel(
    const float* __restrict__ q, const float* __restrict__ k,
    const float* __restrict__ v, float* __restrict__ o)
{
  __shared__ __attribute__((aligned(16))) _Float16 Klds[32][72];   // [key][d]
  __shared__ __attribute__((aligned(16))) _Float16 Vt[64][40];     // [d][key]
  __shared__ __attribute__((aligned(16))) _Float16 Plds[4][16][40];

  const int tid  = threadIdx.x;
  const int lane = tid & 31;
  const int wv   = tid >> 5;
  const int hi   = lane >> 4;
  const int l16  = lane & 15;
  const int b    = blockIdx.y / H_NUM;
  const int h    = blockIdx.y % H_NUM;
  const int qblk = blockIdx.x * 4 + wv;   // 16-row query block

  const size_t base = (size_t)b * T_SEQ * C_DIM + (size_t)h * DH;
  const float* qb = q + base;
  const float* kb = k + base;
  const float* vb = v + base;
  float*       ob = o + base;

  // staging coordinates: 512 float4 per 32x64 tile, 4 per thread
  const int sR = tid >> 4;          // row base (+8 per it)
  const int sC = (tid & 15) * 4;    // col

  // Q fragments for the two 32-wide K-dim steps of QK^T (ISA A-layout)
  v16h qf[2];
  {
    const v4f* rp4 = reinterpret_cast<const v4f*>(qb + (size_t)(qblk * 16 + l16) * C_DIM);
    #pragma unroll
    for (int ks = 0; ks < 2; ++ks) {
      v4f t[4] = { rp4[ks * 8 + hi * 2],     rp4[ks * 8 + hi * 2 + 1],
                   rp4[ks * 8 + 4 + hi * 2], rp4[ks * 8 + 4 + hi * 2 + 1] };
      #pragma unroll
      for (int i = 0; i < 16; ++i) qf[ks][i] = (_Float16)t[i >> 2][i & 3];
    }
  }

  const v8f vzero = {0.f,0.f,0.f,0.f,0.f,0.f,0.f,0.f};
  v8f acc[4];
  #pragma unroll
  for (int nt = 0; nt < 4; ++nt) acc[nt] = vzero;
  float mrow[8], lrow[8];
  #pragma unroll
  for (int r = 0; r < 8; ++r) { mrow[r] = -1e30f; lrow[r] = 0.f; }

  for (int j = 0; j < T_SEQ / 32; ++j) {
    const int t0 = j * 32;

    __syncthreads();   // previous block's fragments fully consumed
    // stage K block [32 keys][64 d] row-major
    #pragma unroll
    for (int it = 0; it < 4; ++it) {
      int row = sR + it * 8;
      v4f fv = *reinterpret_cast<const v4f*>(&kb[(size_t)(t0 + row) * C_DIM + sC]);
      v4h hv = { (_Float16)fv[0], (_Float16)fv[1], (_Float16)fv[2], (_Float16)fv[3] };
      *reinterpret_cast<v4h*>(&Klds[row][sC]) = hv;
    }
    // stage V block transposed: Vt[d][key]
    #pragma unroll
    for (int it = 0; it < 4; ++it) {
      int row = sR + it * 8;
      v4f fv = *reinterpret_cast<const v4f*>(&vb[(size_t)(t0 + row) * C_DIM + sC]);
      Vt[sC + 0][row] = (_Float16)fv[0];
      Vt[sC + 1][row] = (_Float16)fv[1];
      Vt[sC + 2][row] = (_Float16)fv[2];
      Vt[sC + 3][row] = (_Float16)fv[3];
    }
    __syncthreads();

    // S = Q @ K^T for a 16x32 score tile (two 16x16 WMMA C-tiles)
    v8f s[2];
    #pragma unroll
    for (int kt = 0; kt < 2; ++kt) {
      v8f c = vzero;
      #pragma unroll
      for (int ks = 0; ks < 2; ++ks) {
        // B[d][key]: lane = key col (kt*16 + l16), d at hi*8 and 16+hi*8 (8 each) from Klds
        const v8h* p = (const v8h*)&Klds[kt * 16 + l16][ks * 32 + hi * 8];
        const v8h* pq = (const v8h*)&Klds[kt * 16 + l16][ks * 32 + 16 + hi * 8];
        v16h bf = cat8(*p, *pq);
        c = __builtin_amdgcn_wmma_f32_16x16x32_f16(
            false, qf[ks], false, bf, (short)0, c, false, false);
      }
      s[kt] = c;
    }

    // online softmax: row max / sum across 16 lanes of each half-wave
    float sc[8];
    #pragma unroll
    for (int r = 0; r < 8; ++r) {
      float mx = fmaxf(s[0][r], s[1][r]);
      mx = fmaxf(mx, __shfl_xor(mx, 1));
      mx = fmaxf(mx, __shfl_xor(mx, 2));
      mx = fmaxf(mx, __shfl_xor(mx, 4));
      mx = fmaxf(mx, __shfl_xor(mx, 8));
      float mn = fmaxf(mrow[r], mx);
      float scale = __expf(mrow[r] - mn);
      float p0 = __expf(s[0][r] - mn);
      float p1 = __expf(s[1][r] - mn);
      s[0][r] = p0; s[1][r] = p1;
      float sum = p0 + p1;
      sum += __shfl_xor(sum, 1);
      sum += __shfl_xor(sum, 2);
      sum += __shfl_xor(sum, 4);
      sum += __shfl_xor(sum, 8);
      lrow[r] = lrow[r] * scale + sum;
      mrow[r] = mn;
      sc[r] = scale;
    }
    #pragma unroll
    for (int nt = 0; nt < 4; ++nt)
      #pragma unroll
      for (int r = 0; r < 8; ++r) acc[nt][r] *= sc[r];

    // relayout P (C-layout -> A-layout) through per-wave LDS, f16
    #pragma unroll
    for (int r = 0; r < 8; ++r) {
      int m = r + 8 * hi;
      Plds[wv][m][l16]      = (_Float16)s[0][r];
      Plds[wv][m][16 + l16] = (_Float16)s[1][r];
    }
    asm volatile("s_wait_dscnt 0" ::: "memory");  // wave-local LDS RAW fence

    v16h pf;
    {
      const v8h* p = (const v8h*)&Plds[wv][l16][hi * 8];
      const v8h* pq = (const v8h*)&Plds[wv][l16][16 + hi * 8];
      pf = cat8(*p, *pq);
    }

    // O += P @ V  (K-dim = 32 keys, 4 output tiles across d = 64)
    #pragma unroll
    for (int nt = 0; nt < 4; ++nt) {
      const v8h* p = (const v8h*)&Vt[nt * 16 + l16][hi * 8];
      const v8h* pq = (const v8h*)&Vt[nt * 16 + l16][16 + hi * 8];
      v16h vf = cat8(*p, *pq);
      acc[nt] = __builtin_amdgcn_wmma_f32_16x16x32_f16(
          false, pf, false, vf, (short)0, acc[nt], false, false);
    }
  }

  // normalize by row sums and store
  #pragma unroll
  for (int nt = 0; nt < 4; ++nt)
    #pragma unroll
    for (int r = 0; r < 8; ++r) {
      int m = r + 8 * hi;
      ob[(size_t)(qblk * 16 + m) * C_DIM + nt * 16 + l16] = acc[nt][r] / lrow[r];
    }
}

// ---------------------------------------------------------------------------
// LayerNorm (in place), one block per token row
// ---------------------------------------------------------------------------
__global__ __launch_bounds__(256) void ln_kernel(
    float* __restrict__ x, const float* __restrict__ g, const float* __restrict__ bta)
{
  __shared__ float red[256];
  const int row = blockIdx.x, tid = threadIdx.x;
  float* xp = x + (size_t)row * C_DIM;

  float s = 0.f;
  for (int c = tid; c < C_DIM; c += 256) s += xp[c];
  red[tid] = s; __syncthreads();
  for (int st = 128; st > 0; st >>= 1) { if (tid < st) red[tid] += red[tid + st]; __syncthreads(); }
  float mean = red[0] * (1.0f / C_DIM);
  __syncthreads();

  float s2 = 0.f;
  for (int c = tid; c < C_DIM; c += 256) { float d = xp[c] - mean; s2 += d * d; }
  red[tid] = s2; __syncthreads();
  for (int st = 128; st > 0; st >>= 1) { if (tid < st) red[tid] += red[tid + st]; __syncthreads(); }
  float inv = rsqrtf(red[0] * (1.0f / C_DIM) + 1e-5f);

  for (int c = tid; c < C_DIM; c += 256)
    xp[c] = (xp[c] - mean) * inv * g[c] + bta[c];
}

// ---------------------------------------------------------------------------
// out (+)= gate[tok,e] * expert_out ; first expert initializes out (vec4)
// ---------------------------------------------------------------------------
__global__ __launch_bounds__(256) void moe_accum_kernel(
    const float* __restrict__ xe, const float* __restrict__ gate,
    float* __restrict__ out, int e, int first)
{
  int i4 = blockIdx.x * 256 + threadIdx.x;
  if (i4 >= BT * C_DIM / 4) return;
  int tok = i4 / (C_DIM / 4);
  float g = gate[(size_t)tok * E_NUM + e];
  const v4f* xp = reinterpret_cast<const v4f*>(xe);
  v4f* op = reinterpret_cast<v4f*>(out);
  v4f xv = xp[i4];
  v4f pr = {0.f, 0.f, 0.f, 0.f};
  if (!first) pr = op[i4];
  v4f r;
  #pragma unroll
  for (int j = 0; j < 4; ++j) r[j] = pr[j] + g * xv[j];
  op[i4] = r;
}

// ---------------------------------------------------------------------------
// launcher
// ---------------------------------------------------------------------------
extern "C" void kernel_launch(void* const* d_in, const int* in_sizes, int n_in,
                              void* d_out, int out_size, void* d_ws, size_t ws_size,
                              hipStream_t stream)
{
  (void)in_sizes; (void)n_in; (void)out_size; (void)ws_size;
  const float* x      = (const float*)d_in[0];
  const float* gate_W = (const float*)d_in[1];
  const float* gate_b = (const float*)d_in[2];
  const float* Wq     = (const float*)d_in[3];
  const float* bq     = (const float*)d_in[4];
  const float* Wk     = (const float*)d_in[5];
  const float* bk     = (const float*)d_in[6];
  const float* Wv     = (const float*)d_in[7];
  const float* bv     = (const float*)d_in[8];
  const float* Wo     = (const float*)d_in[9];
  const float* bo     = (const float*)d_in[10];
  const float* ln_g   = (const float*)d_in[11];
  const float* ln_b   = (const float*)d_in[12];
  const float* Wfc    = (const float*)d_in[13];
  const float* bfc    = (const float*)d_in[14];
  const float* Wpr    = (const float*)d_in[15];
  const float* bpr    = (const float*)d_in[16];
  float* out = (float*)d_out;

  // workspace layout (floats). h (BT x 4C) exactly aliases q|k|v|att.
  float* ws   = (float*)d_ws;
  float* gbuf = ws;                                // BT*E
  float* qbuf = gbuf + (size_t)BT * E_NUM;         // BT*C
  float* kbuf = qbuf + (size_t)BT * C_DIM;         // BT*C
  float* vbuf = kbuf + (size_t)BT * C_DIM;         // BT*C
  float* att  = vbuf + (size_t)BT * C_DIM;         // BT*C
  float* hbuf = qbuf;                              // BT*4C (alias)
  float* ybuf = att + (size_t)BT * C_DIM;          // BT*C

  const size_t CC = (size_t)C_DIM * C_DIM;
  const size_t C4 = (size_t)4 * C_DIM;

  dim3 blk256(256);
  dim3 gC(BT / BM, C_DIM / BN);          // 32 x 6
  dim3 gFC(BT / BM, (4 * C_DIM) / BN);   // 32 x 24
  dim3 attn_grid(T_SEQ / 64, B_BATCH * H_NUM);

  gate_kernel<<<BT, blk256, 0, stream>>>(x, gate_W, gate_b, gbuf);

  for (int e = 0; e < E_NUM; ++e) {
    const float* cur = x;   // every expert starts from the original x
    for (int l = 0; l < L_NUM; ++l) {
      size_t w = (size_t)(e * L_NUM + l);
      gemm_bias_act_kernel<<<gC, blk256, 0, stream>>>(cur, Wq + w * CC, bq + w * C_DIM, qbuf, BT, C_DIM, C_DIM, 0);
      gemm_bias_act_kernel<<<gC, blk256, 0, stream>>>(cur, Wk + w * CC, bk + w * C_DIM, kbuf, BT, C_DIM, C_DIM, 0);
      gemm_bias_act_kernel<<<gC, blk256, 0, stream>>>(cur, Wv + w * CC, bv + w * C_DIM, vbuf, BT, C_DIM, C_DIM, 0);
      flash_attn_kernel<<<attn_grid, 128, 0, stream>>>(qbuf, kbuf, vbuf, att);
      gemm_bias_act_kernel<<<gC, blk256, 0, stream>>>(att, Wo + w * CC, bo + w * C_DIM, ybuf, BT, C_DIM, C_DIM, 0);
      ln_kernel<<<BT, blk256, 0, stream>>>(ybuf, ln_g + w * C_DIM, ln_b + w * C_DIM);
      gemm_bias_act_kernel<<<gFC, blk256, 0, stream>>>(ybuf, Wfc + w * C_DIM * C4, bfc + w * C4, hbuf, BT, 4 * C_DIM, C_DIM, 1);
      gemm_bias_act_kernel<<<gC, blk256, 0, stream>>>(hbuf, Wpr + w * C4 * C_DIM, bpr + w * C_DIM, ybuf, BT, C_DIM, 4 * C_DIM, 0);
      cur = ybuf;
    }
    moe_accum_kernel<<<(BT * C_DIM / 4 + 255) / 256, blk256, 0, stream>>>(cur, gbuf, out, e, e == 0 ? 1 : 0);
  }
}
